// MAGNN_attn_intra_42820823941457
// MI455X (gfx1250) — matrix operand order repaired and checked by reading, not compiled, not run on hardware
//
#include <hip/hip_runtime.h>

// ---------------------------------------------------------------------------
// MAGNN intra-metapath attention (gfx1250 / MI455X)
//
// Memory-bound kernel set (~1.45 GB total traffic -> ~62 us at 23.3 TB/s):
//   P0 zero  : emax/den scratch + out accumulator
//   P1 score : er[e,h] = leakyrelu(<feat[e,h,:], attn_r[h,:]>) via
//              V_WMMA_F32_16X16X32_F16 (16-edge tile x 1 head per WMMA,
//              B replicated over N columns so extraction only relies on the
//              *documented* C/D layout), + encoded-uint atomic max per dst.
//   P2 den   : den[n,h] = sum exp(e - emax) via global_atomic_add_f32 (L2).
//   P3 agg   : out[n,h,:] += a * feat[e,h,:]; 1 thread per (e,h,d), fully
//              coalesced, nontemporal feat loads (out array ~102 MB stays
//              resident in the 192 MB L2; atomics resolve there).
//   P4 elu   : out = x>0 ? x : expm1(x)
// ---------------------------------------------------------------------------

#define H_ 8
#define D_ 32
#define HD_ 256
#define NEG_SLOPE 0.01f

typedef __attribute__((ext_vector_type(16))) _Float16 v16h;
typedef __attribute__((ext_vector_type(8)))  float    v8f;
typedef __attribute__((ext_vector_type(4)))  float    v4f;

// Order-preserving float <-> uint encoding so we can use integer atomic max.
__device__ __forceinline__ unsigned enc_f32(float f) {
  unsigned u = __float_as_uint(f);
  return (u & 0x80000000u) ? ~u : (u | 0x80000000u);
}
__device__ __forceinline__ float dec_f32(unsigned u) {
  u = (u & 0x80000000u) ? (u & 0x7fffffffu) : ~u;
  return __uint_as_float(u);
}

__global__ void magnn_zero(unsigned* __restrict__ p, long long n) {
  long long i = (long long)blockIdx.x * blockDim.x + threadIdx.x;
  if (i < n) p[i] = 0u;
}

// ---- P1: edge scores via WMMA -------------------------------------------
// One wave handles a tile of 16 consecutive edges; loops over 8 heads.
// A (16x32 f16): documented striping -> lane grp=lane>>4, m=lane&15 supplies
//   edge row (tile*16+m), K slots = {grp*8..grp*8+7, grp*8+16..grp*8+23}.
// B (32x16 f16): attn_r[h,:] replicated across all 16 N-columns, so every
//   output column carries er; per documented C/D layout VGPR p of half-wave
//   grp holds er for edge M = p + 8*grp.
__global__ void magnn_score_wmma(const float* __restrict__ feat0,
                                 const float* __restrict__ attn_r,
                                 const int* __restrict__ dst_idx,
                                 float* __restrict__ er,
                                 unsigned* __restrict__ emax_enc,
                                 int E) {
  __shared__ _Float16 s_attn[HD_];
  int t = threadIdx.x;
  s_attn[t] = (_Float16)attn_r[t];   // blockDim.x == 256 == H_*D_
  __syncthreads();

  int wave = blockIdx.x * (blockDim.x >> 5) + (t >> 5);
  int lane = t & 31;
  int ntiles = (E + 15) >> 4;
  if (wave >= ntiles) return;        // wave-uniform: EXEC all-1 at WMMA

  int e0  = wave << 4;
  int grp = lane >> 4;
  int m   = lane & 15;
  int ea  = e0 + m;                  // edge whose A-rows this lane supplies
  bool have = (ea < E);

  for (int h = 0; h < H_; ++h) {
    // B: value depends only on K (replicated over N); lane group grp holds
    // K = {grp*8..grp*8+7, grp*8+16..grp*8+23}.
    v16h b;
    #pragma unroll
    for (int s = 0; s < 16; ++s)
      b[s] = s_attn[h * D_ + grp * 8 + (s < 8 ? s : s + 8)];

    // A: two contiguous 8-float runs of this edge's head-h features.
    v16h a;
    #pragma unroll
    for (int s = 0; s < 16; ++s) a[s] = (_Float16)0.0f;
    if (have) {
      const float* base = feat0 + (size_t)ea * HD_ + h * D_ + grp * 8;
      v4f r0 = __builtin_nontemporal_load((const v4f*)(base));
      v4f r1 = __builtin_nontemporal_load((const v4f*)(base + 4));
      v4f r2 = __builtin_nontemporal_load((const v4f*)(base + 16));
      v4f r3 = __builtin_nontemporal_load((const v4f*)(base + 20));
      #pragma unroll
      for (int j = 0; j < 4; ++j) {
        a[j]      = (_Float16)r0[j];
        a[4 + j]  = (_Float16)r1[j];
        a[8 + j]  = (_Float16)r2[j];
        a[12 + j] = (_Float16)r3[j];
      }
    }

    v8f c = {};
    c = __builtin_amdgcn_wmma_f32_16x16x32_f16(
        /*neg_a=*/false, a, /*neg_b=*/false, b,
        /*c_mod=*/(short)0, c, /*reuse_a=*/false, /*reuse_b=*/false);

    // Columns replicated -> lane (grp, m<8) extracts edge M = m + 8*grp.
    if (m < 8) {
      int e2 = e0 + grp * 8 + m;
      if (e2 < E) {
        float sc = c[m];
        sc = sc > 0.0f ? sc : NEG_SLOPE * sc;          // leaky relu
        er[(size_t)e2 * H_ + h] = sc;
        int dn = dst_idx[e2];
        __hip_atomic_fetch_max(&emax_enc[(size_t)dn * H_ + h], enc_f32(sc),
                               __ATOMIC_RELAXED, __HIP_MEMORY_SCOPE_AGENT);
      }
    }
  }
}

// ---- P2: softmax denominators -------------------------------------------
__global__ void magnn_den(const float* __restrict__ er,
                          const int* __restrict__ dst_idx,
                          const unsigned* __restrict__ emax_enc,
                          float* __restrict__ den, long long EH) {
  long long idx = (long long)blockIdx.x * blockDim.x + threadIdx.x;
  if (idx >= EH) return;
  int e = (int)(idx >> 3);
  int h = (int)(idx & 7);
  int dn = dst_idx[e];
  float mx = dec_f32(emax_enc[(size_t)dn * H_ + h]);
  float num = expf(er[idx] - mx);
  __hip_atomic_fetch_add(&den[(size_t)dn * H_ + h], num,
                         __ATOMIC_RELAXED, __HIP_MEMORY_SCOPE_AGENT);
}

// ---- P3: attention-weighted scatter aggregation -------------------------
__global__ void magnn_aggregate(const float* __restrict__ feat0,
                                const float* __restrict__ er,
                                const int* __restrict__ dst_idx,
                                const unsigned* __restrict__ emax_enc,
                                const float* __restrict__ den,
                                float* __restrict__ out, long long EHD) {
  long long idx = (long long)blockIdx.x * blockDim.x + threadIdx.x;
  if (idx >= EHD) return;
  int e = (int)(idx >> 8);
  int f = (int)(idx & 255);
  int h = f >> 5;
  int dn = dst_idx[e];
  float sc = er[(size_t)e * H_ + h];
  float mx = dec_f32(emax_enc[(size_t)dn * H_ + h]);
  float dv = den[(size_t)dn * H_ + h];
  float a  = expf(sc - mx) / dv;
  float v  = a * __builtin_nontemporal_load(&feat0[idx]);
  __hip_atomic_fetch_add(&out[(size_t)dn * HD_ + f], v,
                         __ATOMIC_RELAXED, __HIP_MEMORY_SCOPE_AGENT);
}

// ---- P4: ELU -------------------------------------------------------------
__global__ void magnn_elu(float* __restrict__ out, long long n) {
  long long i = (long long)blockIdx.x * blockDim.x + threadIdx.x;
  if (i >= n) return;
  float x = out[i];
  out[i] = x > 0.0f ? x : expm1f(x);
}

extern "C" void kernel_launch(void* const* d_in, const int* in_sizes, int n_in,
                              void* d_out, int out_size, void* d_ws, size_t ws_size,
                              hipStream_t stream) {
  const float* feat0  = (const float*)d_in[0];
  const float* attn_r = (const float*)d_in[1];
  const int*   dst    = (const int*)d_in[2];
  float*       out    = (float*)d_out;

  const int E = in_sizes[0] / HD_;
  const int N = out_size / HD_;

  // Workspace layout: [er: E*H f32][emax_enc: N*H u32][den: N*H f32] ~22.4 MB
  float*    er       = (float*)d_ws;
  unsigned* emax_enc = (unsigned*)((char*)d_ws + (size_t)E * H_ * sizeof(float));
  float*    den      = (float*)((char*)emax_enc + (size_t)N * H_ * sizeof(unsigned));

  const long long NH  = (long long)N * H_;
  const long long EH  = (long long)E * H_;
  const long long EHD = (long long)E * HD_;
  const long long NHD = (long long)N * HD_;

  // P0: zero emax_enc+den (contiguous) and out accumulator.
  magnn_zero<<<(unsigned)((2 * NH + 255) / 256), 256, 0, stream>>>(emax_enc, 2 * NH);
  magnn_zero<<<(unsigned)((NHD + 255) / 256), 256, 0, stream>>>((unsigned*)out, NHD);

  // P1: scores (1 wave per 16-edge tile, 8 waves / 256-thread block).
  int ntiles = (E + 15) / 16;
  int blocks = (ntiles + 7) / 8;
  magnn_score_wmma<<<blocks, 256, 0, stream>>>(feat0, attn_r, dst, er, emax_enc, E);

  // P2: denominators.
  magnn_den<<<(unsigned)((EH + 255) / 256), 256, 0, stream>>>(er, dst, emax_enc, den, EH);

  // P3: aggregation.
  magnn_aggregate<<<(unsigned)((EHD + 255) / 256), 256, 0, stream>>>(
      feat0, er, dst, emax_enc, den, out, EHD);

  // P4: ELU.
  magnn_elu<<<(unsigned)((NHD + 255) / 256), 256, 0, stream>>>(out, NHD);
}
